// TimeAgnosticWindShooterLoss_80204219285757
// MI455X (gfx1250) — matrix operand order repaired and checked by reading, not compiled, not run on hardware
//
#include <hip/hip_runtime.h>
#include <hip/hip_bf16.h>
#include <math.h>

// Problem dims (match reference)
#define B_    4096
#define DIN_  2048
#define H_    1024
#define D_    256
#define HW_   256
#define STEPS_ 15

#define MB_ ((size_t)1 << 20)

typedef __attribute__((ext_vector_type(16))) _Float16 v16h;
typedef __attribute__((ext_vector_type(8)))  _Float16 v8h;
typedef __attribute__((ext_vector_type(8)))  float    v8f;

__device__ __forceinline__ float wave_sum(float s) {
#pragma unroll
  for (int o = 16; o > 0; o >>= 1) s += __shfl_xor(s, o, 32);
  return s;
}

// ---------------- converts ----------------
__global__ __launch_bounds__(256) void cvt_f16(const float* __restrict__ s,
                                               _Float16* __restrict__ d, size_t n) {
  size_t i = (size_t)blockIdx.x * blockDim.x + threadIdx.x;
  if (i < n) d[i] = (_Float16)s[i];
}

__global__ __launch_bounds__(256) void cvtT_f16(const float* __restrict__ s,
                                                _Float16* __restrict__ d, int R, int C) {
  int i = blockIdx.x * blockDim.x + threadIdx.x;
  if (i < R * C) {
    int r = i / C, c = i % C;
    d[(size_t)c * R + r] = (_Float16)s[i];
  }
}

// ---------------- WMMA GEMM fragments ----------------
union AU { v16h v; v8h h[2]; };
struct Frag { AU a; v16h b[4]; };

__device__ __forceinline__ void load_frag(Frag& f, const _Float16* __restrict__ arow,
                                          const _Float16* __restrict__ brow, int ldb) {
  // A fragment (16x32 f16): lane holds row; Ks = [8h..8h+7] ++ [16+8h..+7]
  f.a.h[0] = *(const v8h*)(arow);
  f.a.h[1] = *(const v8h*)(arow + 16);
  // B fragments (32x16 f16): lane holds N-col; Ks = [8h..8h+7] ++ [16+8h..+7]
#pragma unroll
  for (int t = 0; t < 4; ++t)
#pragma unroll
    for (int s = 0; s < 16; ++s)
      f.b[t][s] = brow[(size_t)(s < 8 ? s : s + 8) * ldb + t * 16];
}

__device__ __forceinline__ void mma4(v8f acc[4], const Frag& f) {
  // 4 back-to-back WMMAs sharing A: reuse_a hint on the first three.
  acc[0] = __builtin_amdgcn_wmma_f32_16x16x32_f16(
      false, f.a.v, false, f.b[0], (short)0, acc[0], true,  false);
  acc[1] = __builtin_amdgcn_wmma_f32_16x16x32_f16(
      false, f.a.v, false, f.b[1], (short)0, acc[1], true,  false);
  acc[2] = __builtin_amdgcn_wmma_f32_16x16x32_f16(
      false, f.a.v, false, f.b[2], (short)0, acc[2], true,  false);
  acc[3] = __builtin_amdgcn_wmma_f32_16x16x32_f16(
      false, f.a.v, false, f.b[3], (short)0, acc[3], false, false);
}

// ---------------- WMMA GEMM ----------------
// C[M,N] = act(A[M,K] @ Bw[K,N] + bias); requires K % 64 == 0.
// ACT: 0 = identity, 1 = tanh, 2 = (1 - tanh^2) * wp2[n]  (grad-phi weights)
// Block = 128 threads (4 waves); wave w computes rows [blockIdx.y*64 + 16w, +16),
// cols [blockIdx.x*64, +64) as 4 WMMA N-tiles.
// K-loop unrolled x2 with ping-pong fragment registers: no inter-buffer moves,
// no WMMA->VALU hazard NOPs, loads a full chunk ahead of the matrix pipe.
template<int ACT, bool OUTF, bool OUTH>
__global__ __launch_bounds__(128) void gemm_wmma(
    const _Float16* __restrict__ A, const _Float16* __restrict__ Bw,
    const float* __restrict__ bias, const float* __restrict__ wp2,
    float* __restrict__ Cf, _Float16* __restrict__ Ch, int M, int N, int K)
{
  const int lane = threadIdx.x & 31;
  const int wave = threadIdx.x >> 5;
  const int hl   = lane >> 4;   // lane half-group
  const int rc   = lane & 15;   // row (A) / col (C,D) index
  const int m0   = (blockIdx.y * 4 + wave) * 16;
  const int n0   = blockIdx.x * 64;

  const v8f zero = {0.f, 0.f, 0.f, 0.f, 0.f, 0.f, 0.f, 0.f};
  v8f acc[4] = {zero, zero, zero, zero};

  const _Float16* ap = A + (size_t)(m0 + rc) * K + hl * 8;
  const _Float16* bp = Bw + (size_t)(hl * 8) * N + n0 + rc;
  const size_t bstep = (size_t)32 * N;

  Frag f0, f1;
  load_frag(f0, ap, bp, N);                    // k = 0
  for (int k = 0; k < K; k += 64) {
    load_frag(f1, ap + 32, bp + bstep, N);     // k + 32 (always < K: K%64==0)
    __builtin_prefetch(ap + 64, 0, 3);
    mma4(acc, f0);
    if (k + 64 < K) load_frag(f0, ap + 64, bp + 2 * bstep, N);
    mma4(acc, f1);
    ap += 64;
    bp += 2 * bstep;
  }

#pragma unroll
  for (int t = 0; t < 4; ++t) {
    const int n = n0 + t * 16 + rc;
    const float bs = bias ? bias[n] : 0.0f;
    float w2 = 0.0f;
    if (ACT == 2) w2 = wp2[n];
#pragma unroll
    for (int r = 0; r < 8; ++r) {
      const int m = m0 + hl * 8 + r;
      float val = acc[t][r] + bs;
      if (ACT == 1) val = tanhf(val);
      if (ACT == 2) { const float u = tanhf(val); val = (1.0f - u * u) * w2; }
      if (OUTF) Cf[(size_t)m * N + n] = val;
      if (OUTH) Ch[(size_t)m * N + n] = (_Float16)val;
    }
  }
}

// ---------------- trace state init ----------------
// One wave32 per row. Applies the degenerate-wind noise fallback, seeds
// x = z_s, v = vs = W_start, xh = f16(z_s), min_d = +inf.
__global__ __launch_bounds__(256) void init_state(
    const float* __restrict__ z_s, const float* __restrict__ wst,
    const float* __restrict__ noise, float* __restrict__ x, float* __restrict__ v,
    float* __restrict__ vs, _Float16* __restrict__ xh, float* __restrict__ min_d)
{
  const int lane = threadIdx.x & 31;
  const int row  = blockIdx.x * 8 + (threadIdx.x >> 5);
  const size_t base = (size_t)row * D_;
  float wv[8], nv[8], zv[8];
  float wn2 = 0.f, nn2 = 0.f;
#pragma unroll
  for (int j = 0; j < 8; ++j) {
    const size_t idx = base + j * 32 + lane;
    wv[j] = wst[idx]; nv[j] = noise[idx]; zv[j] = z_s[idx];
    wn2 += wv[j] * wv[j]; nn2 += nv[j] * nv[j];
  }
  wn2 = wave_sum(wn2); nn2 = wave_sum(nn2);
  const float wn = sqrtf(wn2 + 1e-24f);
  const float ns = 1e-4f / (sqrtf(nn2 + 1e-24f) + 1e-12f);
  const bool deg = wn < 1e-5f;
#pragma unroll
  for (int j = 0; j < 8; ++j) {
    const size_t idx = base + j * 32 + lane;
    const float w = wv[j] + (deg ? nv[j] * ns : 0.0f);
    v[idx] = w; vs[idx] = w; x[idx] = zv[j]; xh[idx] = (_Float16)zv[j];
  }
  if (lane == 0) min_d[row] = 3.4e38f;
}

// ---------------- RK4 stage combine ----------------
// a = -(g.vs)vs + 0.5(vs.vs)g ; accumulate Butcher sums; produce next eval
// point (xh f16, vs) or, on LAST, commit xn/vn and update running min dist^2.
template<bool FIRST, bool LAST>
__global__ __launch_bounds__(256) void rk_stage(
    const float* __restrict__ g, float* __restrict__ x, float* __restrict__ v,
    float* __restrict__ vs, _Float16* __restrict__ xh,
    float* __restrict__ sum_x, float* __restrict__ sum_v,
    const float* __restrict__ z_e, float* __restrict__ min_d,
    float c, float w, float h6)
{
  const int lane = threadIdx.x & 31;
  const int row  = blockIdx.x * 8 + (threadIdx.x >> 5);
  const size_t base = (size_t)row * D_;
  float gv8[8], vs8[8];
  float gv = 0.f, vv = 0.f;
#pragma unroll
  for (int j = 0; j < 8; ++j) {
    const size_t idx = base + j * 32 + lane;
    gv8[j] = g[idx]; vs8[j] = vs[idx];
    gv += gv8[j] * vs8[j]; vv += vs8[j] * vs8[j];
  }
  gv = wave_sum(gv); vv = wave_sum(vv);
  float dd = 0.f;
#pragma unroll
  for (int j = 0; j < 8; ++j) {
    const size_t idx = base + j * 32 + lane;
    const float a = -gv * vs8[j] + 0.5f * vv * gv8[j];
    if (LAST) {
      const float sxf = sum_x[idx] + w * vs8[j];
      const float svf = sum_v[idx] + w * a;
      const float xn  = x[idx] + h6 * sxf;
      const float vn  = v[idx] + h6 * svf;
      x[idx] = xn; v[idx] = vn; vs[idx] = vn; xh[idx] = (_Float16)xn;
      const float dz = xn - z_e[idx];
      dd += dz * dz;
    } else {
      if (FIRST) { sum_x[idx] = vs8[j];          sum_v[idx] = a; }
      else       { sum_x[idx] += w * vs8[j];     sum_v[idx] += w * a; }
      xh[idx] = (_Float16)(x[idx] + c * vs8[j]);  // next eval point uses CURRENT vs
      vs[idx] = v[idx] + c * a;
    }
  }
  if (LAST) {
    dd = wave_sum(dd);
    if (lane == 0) min_d[row] = fminf(min_d[row], dd);
  }
}

// ---------------- final mean (single block: deterministic) ----------------
__global__ __launch_bounds__(256) void reduce_mean(const float* __restrict__ md,
                                                   float* __restrict__ out, int n) {
  __shared__ float sm[256];
  float s = 0.f;
  for (int i = threadIdx.x; i < n; i += 256) s += md[i];
  sm[threadIdx.x] = s;
  __syncthreads();
  for (int o = 128; o > 0; o >>= 1) {
    if ((int)threadIdx.x < o) sm[threadIdx.x] += sm[threadIdx.x + o];
    __syncthreads();
  }
  if (threadIdx.x == 0) out[0] = sm[0] * (1.0f / n);  // WEIGHT = 1.0
}

// ---------------- host orchestration ----------------
extern "C" void kernel_launch(void* const* d_in, const int* in_sizes, int n_in,
                              void* d_out, int out_size, void* d_ws, size_t ws_size,
                              hipStream_t stream) {
  (void)in_sizes; (void)n_in; (void)out_size; (void)ws_size;
  const float* x_start = (const float*)d_in[0];
  const float* x_end   = (const float*)d_in[1];
  const float* noise   = (const float*)d_in[2];
  const float* W1  = (const float*)d_in[3];  const float* b1  = (const float*)d_in[4];
  const float* W2  = (const float*)d_in[5];  const float* b2  = (const float*)d_in[6];
  const float* Ww1 = (const float*)d_in[7];  const float* bw1 = (const float*)d_in[8];
  const float* Ww2 = (const float*)d_in[9];  const float* bw2 = (const float*)d_in[10];
  const float* Wp1 = (const float*)d_in[11]; const float* bp1 = (const float*)d_in[12];
  const float* wp2 = (const float*)d_in[13];

  char* wsp = (char*)d_ws;
  size_t off = 0;
  auto alloc = [&](size_t bytes) -> void* {
    off = (off + 255) & ~(size_t)255;
    void* p = wsp + off;
    off += bytes;
    return p;
  };

  // Big f16 activation regions; re-used (aliased) by trace-phase f32 state
  // once the encoder is done.
  _Float16* Xh = (_Float16*)alloc((size_t)B_ * DIN_ * 2);  // 16 MB
  _Float16* Hh = (_Float16*)alloc((size_t)B_ * H_ * 2);    //  8 MB
  float* x     = (float*)Xh;                               // +0 MB  (4 MB)
  float* v     = (float*)((char*)Xh + 4 * MB_);            // +4 MB
  float* vs    = (float*)((char*)Xh + 8 * MB_);            // +8 MB
  float* g     = (float*)((char*)Xh + 12 * MB_);           // +12 MB (also wind out)
  float* wst   = g;
  float* sum_x = (float*)Hh;
  float* sum_v = (float*)((char*)Hh + 4 * MB_);

  _Float16* W1h   = (_Float16*)alloc((size_t)DIN_ * H_ * 2);
  _Float16* W2h   = (_Float16*)alloc((size_t)H_ * D_ * 2);
  _Float16* Ww1h  = (_Float16*)alloc((size_t)D_ * HW_ * 2);
  _Float16* Ww2h  = (_Float16*)alloc((size_t)HW_ * D_ * 2);
  _Float16* Wp1h  = (_Float16*)alloc((size_t)D_ * HW_ * 2);
  _Float16* Wp1Th = (_Float16*)alloc((size_t)HW_ * D_ * 2);
  float*    z_s   = (float*)alloc((size_t)B_ * D_ * 4);
  float*    z_e   = (float*)alloc((size_t)B_ * D_ * 4);
  _Float16* zsh   = (_Float16*)alloc((size_t)B_ * D_ * 2);
  _Float16* xh    = (_Float16*)alloc((size_t)B_ * D_ * 2);
  _Float16* hb    = (_Float16*)alloc((size_t)B_ * HW_ * 2);
  float*    min_d = (float*)alloc((size_t)B_ * 4);

  const int TPB = 256;
  auto nblk = [](size_t n) { return (unsigned)((n + 255) / 256); };
  const dim3 GB(128);

  // Weight converts (f16, plus transposed Wp1 for the grad-phi backward GEMM)
  cvt_f16<<<nblk((size_t)DIN_ * H_), TPB, 0, stream>>>(W1, W1h, (size_t)DIN_ * H_);
  cvt_f16<<<nblk((size_t)H_ * D_), TPB, 0, stream>>>(W2, W2h, (size_t)H_ * D_);
  cvt_f16<<<nblk((size_t)D_ * HW_), TPB, 0, stream>>>(Ww1, Ww1h, (size_t)D_ * HW_);
  cvt_f16<<<nblk((size_t)HW_ * D_), TPB, 0, stream>>>(Ww2, Ww2h, (size_t)HW_ * D_);
  cvt_f16<<<nblk((size_t)D_ * HW_), TPB, 0, stream>>>(Wp1, Wp1h, (size_t)D_ * HW_);
  cvtT_f16<<<nblk((size_t)D_ * HW_), TPB, 0, stream>>>(Wp1, Wp1Th, D_, HW_);

  // Encode x_start -> z_s (f32 + f16 copy for wind)
  cvt_f16<<<nblk((size_t)B_ * DIN_), TPB, 0, stream>>>(x_start, Xh, (size_t)B_ * DIN_);
  gemm_wmma<1, false, true><<<dim3(H_ / 64, B_ / 64), GB, 0, stream>>>(
      Xh, W1h, b1, nullptr, nullptr, Hh, B_, H_, DIN_);
  gemm_wmma<0, true, true><<<dim3(D_ / 64, B_ / 64), GB, 0, stream>>>(
      Hh, W2h, b2, nullptr, z_s, zsh, B_, D_, H_);

  // Encode x_end -> z_e (re-uses Xh/Hh)
  cvt_f16<<<nblk((size_t)B_ * DIN_), TPB, 0, stream>>>(x_end, Xh, (size_t)B_ * DIN_);
  gemm_wmma<1, false, true><<<dim3(H_ / 64, B_ / 64), GB, 0, stream>>>(
      Xh, W1h, b1, nullptr, nullptr, Hh, B_, H_, DIN_);
  gemm_wmma<0, true, false><<<dim3(D_ / 64, B_ / 64), GB, 0, stream>>>(
      Hh, W2h, b2, nullptr, z_e, nullptr, B_, D_, H_);

  // Wind field at z_s (Xh region is now free -> wst aliases g)
  gemm_wmma<1, false, true><<<dim3(HW_ / 64, B_ / 64), GB, 0, stream>>>(
      zsh, Ww1h, bw1, nullptr, nullptr, hb, B_, HW_, D_);
  gemm_wmma<0, true, false><<<dim3(D_ / 64, B_ / 64), GB, 0, stream>>>(
      hb, Ww2h, bw2, nullptr, wst, nullptr, B_, D_, HW_);

  init_state<<<B_ / 8, TPB, 0, stream>>>(z_s, wst, noise, x, v, vs, xh, min_d);

  // RK4 geodesic trace: 60 dependent grad-phi GEMM pairs + row-wise combines
  const float h = 1.0f / (float)STEPS_;
  for (int s = 0; s < STEPS_; ++s) {
    // stage 1
    gemm_wmma<2, false, true><<<dim3(HW_ / 64, B_ / 64), GB, 0, stream>>>(
        xh, Wp1h, bp1, wp2, nullptr, hb, B_, HW_, D_);
    gemm_wmma<0, true, false><<<dim3(D_ / 64, B_ / 64), GB, 0, stream>>>(
        hb, Wp1Th, nullptr, nullptr, g, nullptr, B_, D_, HW_);
    rk_stage<true, false><<<B_ / 8, TPB, 0, stream>>>(
        g, x, v, vs, xh, sum_x, sum_v, nullptr, nullptr, 0.5f * h, 1.0f, 0.0f);
    // stage 2
    gemm_wmma<2, false, true><<<dim3(HW_ / 64, B_ / 64), GB, 0, stream>>>(
        xh, Wp1h, bp1, wp2, nullptr, hb, B_, HW_, D_);
    gemm_wmma<0, true, false><<<dim3(D_ / 64, B_ / 64), GB, 0, stream>>>(
        hb, Wp1Th, nullptr, nullptr, g, nullptr, B_, D_, HW_);
    rk_stage<false, false><<<B_ / 8, TPB, 0, stream>>>(
        g, x, v, vs, xh, sum_x, sum_v, nullptr, nullptr, 0.5f * h, 2.0f, 0.0f);
    // stage 3
    gemm_wmma<2, false, true><<<dim3(HW_ / 64, B_ / 64), GB, 0, stream>>>(
        xh, Wp1h, bp1, wp2, nullptr, hb, B_, HW_, D_);
    gemm_wmma<0, true, false><<<dim3(D_ / 64, B_ / 64), GB, 0, stream>>>(
        hb, Wp1Th, nullptr, nullptr, g, nullptr, B_, D_, HW_);
    rk_stage<false, false><<<B_ / 8, TPB, 0, stream>>>(
        g, x, v, vs, xh, sum_x, sum_v, nullptr, nullptr, h, 2.0f, 0.0f);
    // stage 4: commit xn/vn, track closest approach
    gemm_wmma<2, false, true><<<dim3(HW_ / 64, B_ / 64), GB, 0, stream>>>(
        xh, Wp1h, bp1, wp2, nullptr, hb, B_, HW_, D_);
    gemm_wmma<0, true, false><<<dim3(D_ / 64, B_ / 64), GB, 0, stream>>>(
        hb, Wp1Th, nullptr, nullptr, g, nullptr, B_, D_, HW_);
    rk_stage<false, true><<<B_ / 8, TPB, 0, stream>>>(
        g, x, v, vs, xh, sum_x, sum_v, z_e, min_d, 0.0f, 1.0f, h / 6.0f);
  }

  reduce_mean<<<1, TPB, 0, stream>>>(min_d, (float*)d_out, B_);
}
